// ContextQueryAttention_64965675319413
// MI455X (gfx1250) — matrix operand order repaired and checked by reading, not compiled, not run on hardware
//
#include <hip/hip_runtime.h>
#include <hip/hip_bf16.h>

// Problem constants (B, DM, M, N from the reference)
constexpr int CB = 16;
constexpr int CD = 128;
constexpr int CM = 512;
constexpr int CN = 4096;

typedef __attribute__((ext_vector_type(16))) __bf16 bf16x16;
typedef __attribute__((ext_vector_type(8)))  __bf16 bf16x8;
typedef __attribute__((ext_vector_type(8)))  float  f32x8;

static __device__ __forceinline__ bf16x16 cat16(bf16x8 a, bf16x8 b) {
  return __builtin_shufflevector(a, b, 0,1,2,3,4,5,6,7,8,9,10,11,12,13,14,15);
}
static __device__ __forceinline__ f32x8 zero8() {
  f32x8 v;
#pragma unroll
  for (int i = 0; i < 8; ++i) v[i] = 0.0f;
  return v;
}
static __device__ __forceinline__ f32x8 wmma_bf16(bf16x16 a, bf16x16 b, f32x8 c) {
  // (neg_a, A, neg_b, B, c_mod, C, reuse_a, reuse_b)
  return __builtin_amdgcn_wmma_f32_16x16x32_bf16(false, a, false, b, (short)0, c, false, false);
}

// ---------------------------------------------------------------------------
// Pass 1a: convert X[b, D, L] -> Xh (bf16, same layout) and XT (bf16, [b, L, D]).
// Optional per-d scale (w3) applied to the transposed copy only.
// grid (L/32, D/32, B), block (32, 8)
// ---------------------------------------------------------------------------
__global__ void k_prep(const float* __restrict__ X, const float* __restrict__ sc,
                       __bf16* __restrict__ Xh, __bf16* __restrict__ XT, int L) {
  __shared__ float tile[32][33];
  const int b  = blockIdx.z;
  const int l0 = blockIdx.x * 32;
  const int d0 = blockIdx.y * 32;
  const int tx = threadIdx.x, ty = threadIdx.y;
#pragma unroll
  for (int i = 0; i < 4; ++i) {
    const int d = d0 + ty + i * 8;
    const int l = l0 + tx;
    const size_t idx = ((size_t)b * CD + d) * L + l;
    const float v = X[idx];
    Xh[idx] = (__bf16)v;
    const float s = sc ? sc[d] : 1.0f;
    tile[ty + i * 8][tx] = v * s;
  }
  __syncthreads();
#pragma unroll
  for (int i = 0; i < 4; ++i) {
    const int l = l0 + ty + i * 8;
    const int d = d0 + tx;
    XT[((size_t)b * L + l) * CD + d] = (__bf16)tile[tx][ty + i * 8];
  }
}

// ---------------------------------------------------------------------------
// Pass 1b: out[b, j] = sum_d w[d] * X[b, d, j].  grid (B*L/256), block 256.
// ---------------------------------------------------------------------------
__global__ void k_rowvec(const float* __restrict__ X, const float* __restrict__ w,
                         float* __restrict__ out, int L) {
  const int idx = blockIdx.x * 256 + threadIdx.x;
  const int b = idx / L, j = idx % L;
  const float* p = X + (size_t)b * CD * L + j;
  float acc = 0.0f;
#pragma unroll 4
  for (int d = 0; d < CD; ++d) acc += w[d] * p[(size_t)d * L];
  out[idx] = acc;
}

// ---------------------------------------------------------------------------
// Pass 2: s[b, n, m] = sum_d (w3[d]*c[b,d,n]) * q[b,d,m] + v1[b,m] + v2[b,n]
// A = cw3T[n, d] (lane = n-row), B = qT[m, d] (lane = m-col), K = d = 128.
// grid (N/16, B), block 256 (8 waves x 64 m each).
// ---------------------------------------------------------------------------
__global__ void __launch_bounds__(256) k_sgemm(
    const __bf16* __restrict__ CT, const __bf16* __restrict__ QT,
    const float* __restrict__ V1, const float* __restrict__ V2,
    float* __restrict__ S) {
  const int b  = blockIdx.y;
  const int n0 = blockIdx.x * 16;
  const int w  = threadIdx.x >> 5, lane = threadIdx.x & 31;
  const int col = lane & 15, hf = lane >> 4;
  const int m0 = w * 64;

  f32x8 acc[4];
#pragma unroll
  for (int t = 0; t < 4; ++t) acc[t] = zero8();

  const __bf16* aRow = CT + ((size_t)b * CN + (n0 + col)) * CD;
#pragma unroll
  for (int kb = 0; kb < CD; kb += 32) {
    // A fragment: rows = n, K interleave {0..7,16..23} / {8..15,24..31}
    const bf16x8 alo = *(const bf16x8*)(aRow + kb + hf * 8);
    const bf16x8 ahi = *(const bf16x8*)(aRow + kb + 16 + hf * 8);
    const bf16x16 A = cat16(alo, ahi);
#pragma unroll
    for (int t = 0; t < 4; ++t) {
      const int m = m0 + t * 16 + col;
      // B fragment: cols = m, K interleave {0..7,16..23} / {8..15,24..31}
      const __bf16* bp = QT + ((size_t)b * CM + m) * CD + kb + hf * 8;
      const bf16x16 Bm = cat16(*(const bf16x8*)bp, *(const bf16x8*)(bp + 16));
      acc[t] = wmma_bf16(A, Bm, acc[t]);
    }
  }
#pragma unroll
  for (int t = 0; t < 4; ++t) {
    const int m = m0 + t * 16 + col;
    const float v1 = V1[(size_t)b * CM + m];
#pragma unroll
    for (int r = 0; r < 8; ++r) {
      const int n = n0 + r + hf * 8;
      S[((size_t)b * CN + n) * CM + m] = acc[t][r] + v1 + V2[(size_t)b * CN + n];
    }
  }
}

// ---------------------------------------------------------------------------
// Pass 3a: row softmax stats (over m, 512 wide). One wave per row.
// grid (B*N/8), block 256.
// ---------------------------------------------------------------------------
__global__ void k_rowstats(const float* __restrict__ S, float* __restrict__ rmax,
                           float* __restrict__ rrs) {
  const int w = threadIdx.x >> 5, lane = threadIdx.x & 31;
  const int row = blockIdx.x * 8 + w;  // row = b*N + n
  const float* p = S + (size_t)row * CM;
  float v[16];
  float mx = -1e30f;
#pragma unroll
  for (int i = 0; i < 16; ++i) { v[i] = p[lane + i * 32]; mx = fmaxf(mx, v[i]); }
  for (int o = 16; o; o >>= 1) mx = fmaxf(mx, __shfl_xor(mx, o, 32));
  float sm = 0.0f;
#pragma unroll
  for (int i = 0; i < 16; ++i) sm += __expf(v[i] - mx);
  for (int o = 16; o; o >>= 1) sm += __shfl_xor(sm, o, 32);
  if (lane == 0) { rmax[row] = mx; rrs[row] = 1.0f / sm; }
}

// ---------------------------------------------------------------------------
// Pass 3b: column softmax partials (over n, split into 8 chunks of 512).
// grid (M/256, 8, B), block 256 — online (max, sum) per column.
// ---------------------------------------------------------------------------
__global__ void k_colpart(const float* __restrict__ S, float* __restrict__ pm,
                          float* __restrict__ ps) {
  const int m  = blockIdx.x * 256 + threadIdx.x;
  const int nc = blockIdx.y;
  const int b  = blockIdx.z;
  const float* p = S + ((size_t)b * CN + nc * 512) * CM + m;
  float mx = -1e30f, sm = 0.0f;
  for (int n = 0; n < 512; ++n) {
    const float x = p[(size_t)n * CM];
    if (x > mx) { sm *= __expf(mx - x); mx = x; }
    sm += __expf(x - mx);
  }
  const size_t o = ((size_t)b * 8 + nc) * CM + m;
  pm[o] = mx; ps[o] = sm;
}

__global__ void k_colmerge(const float* __restrict__ pm, const float* __restrict__ ps,
                           float* __restrict__ cmax, float* __restrict__ crs) {
  const int idx = blockIdx.x * 256 + threadIdx.x;  // b*M + m
  const int b = idx / CM, m = idx % CM;
  float mx = -1e30f;
#pragma unroll
  for (int k = 0; k < 8; ++k) mx = fmaxf(mx, pm[((size_t)b * 8 + k) * CM + m]);
  float sm = 0.0f;
#pragma unroll
  for (int k = 0; k < 8; ++k) {
    const size_t o = ((size_t)b * 8 + k) * CM + m;
    sm += ps[o] * __expf(pm[o] - mx);
  }
  cmax[idx] = mx; crs[idx] = 1.0f / sm;
}

// ---------------------------------------------------------------------------
// Pass 4: tmp[b, d, m] = sum_n sCol[b,n,m] * c[b,d,n].  K = n = 4096.
// A = c[d, n] (native bf16), B = sCol staged in LDS transposed so each lane
// reads two runs of 8 contiguous bf16. grid (M/64, B), block 256 (wave w -> d-tile w).
// ---------------------------------------------------------------------------
__global__ void __launch_bounds__(256) k_tmpgemm(
    const __bf16* __restrict__ CH, const float* __restrict__ S,
    const float* __restrict__ CMAX, const float* __restrict__ CRS,
    __bf16* __restrict__ TMPH) {
  __shared__ __bf16 Bt[64 * 32];  // [m_local][n_local]
  const int b  = blockIdx.y;
  const int m0 = blockIdx.x * 64;
  const int w  = threadIdx.x >> 5, lane = threadIdx.x & 31;
  const int col = lane & 15, hf = lane >> 4;
  const int d0 = w * 16;

  f32x8 acc[4];
#pragma unroll
  for (int t = 0; t < 4; ++t) acc[t] = zero8();

  // staging indices: thread -> (n-row, 8 consecutive m)
  const int nr = threadIdx.x >> 3;
  const int ms = (threadIdx.x & 7) * 8;
  float cm[8], cr[8];
#pragma unroll
  for (int j = 0; j < 8; ++j) {
    const int m = m0 + ms + j;
    cm[j] = CMAX[(size_t)b * CM + m];
    cr[j] = CRS[(size_t)b * CM + m];
  }

  const __bf16* aRow = CH + ((size_t)b * CD + d0 + col) * CN;
  for (int kb = 0; kb < CN; kb += 32) {
    __syncthreads();
    const float* sp = S + ((size_t)b * CN + kb + nr) * CM + m0 + ms;
#pragma unroll
    for (int j = 0; j < 8; ++j) {
      const float v = __expf(sp[j] - cm[j]) * cr[j];
      Bt[(ms + j) * 32 + nr] = (__bf16)v;  // transposed store
    }
    __syncthreads();

    const bf16x8 alo = *(const bf16x8*)(aRow + kb + hf * 8);
    const bf16x8 ahi = *(const bf16x8*)(aRow + kb + 16 + hf * 8);
    const bf16x16 A = cat16(alo, ahi);
#pragma unroll
    for (int t = 0; t < 4; ++t) {
      const __bf16* bp = &Bt[(t * 16 + col) * 32 + hf * 8];
      const bf16x16 Bm = cat16(*(const bf16x8*)bp, *(const bf16x8*)(bp + 16));
      acc[t] = wmma_bf16(A, Bm, acc[t]);
    }
  }
#pragma unroll
  for (int t = 0; t < 4; ++t)
#pragma unroll
    for (int r = 0; r < 8; ++r) {
      const int d = d0 + r + hf * 8;
      const int m = m0 + t * 16 + col;
      TMPH[((size_t)b * CD + d) * CM + m] = (__bf16)acc[t][r];
    }
}

// ---------------------------------------------------------------------------
// Pass 5: a[d,n] = sum_m sRow[n,m] q[d,m]; bb[d,n] = sum_m sRow[n,m] tmp[d,m];
// fused epilogue writes [c, a, c*a, c*bb]. sRow strip (16 x 512) staged once
// in LDS as bf16; the B fragment is shared by both WMMAs per K-step.
// grid (N/16, B), block 256 (wave w -> d-tile w).
// ---------------------------------------------------------------------------
__global__ void __launch_bounds__(256) k_final(
    const __bf16* __restrict__ QH, const __bf16* __restrict__ TMPH,
    const float* __restrict__ S, const float* __restrict__ RMAX,
    const float* __restrict__ RRS, const float* __restrict__ C,
    float* __restrict__ OUT) {
  __shared__ __bf16 sR[16 * CM];
  const int b  = blockIdx.y;
  const int n0 = blockIdx.x * 16;
  const int w  = threadIdx.x >> 5, lane = threadIdx.x & 31;
  const int col = lane & 15, hf = lane >> 4;
  const int d0 = w * 16;

  {  // stage sRow strip: 16 rows x 512 m
    const int nr = threadIdx.x >> 4;
    const int ms = (threadIdx.x & 15) * 32;
    const size_t row = (size_t)b * CN + n0 + nr;
    const float mx = RMAX[row], rs = RRS[row];
    const float* sp = S + row * CM + ms;
#pragma unroll
    for (int j = 0; j < 32; ++j)
      sR[nr * CM + ms + j] = (__bf16)(__expf(sp[j] - mx) * rs);
  }
  __syncthreads();

  f32x8 aa = zero8(), ab = zero8();
  const __bf16* qRow = QH + ((size_t)b * CD + d0 + col) * CM;
  const __bf16* tRow = TMPH + ((size_t)b * CD + d0 + col) * CM;
#pragma unroll
  for (int kb = 0; kb < CM; kb += 32) {
    const __bf16* bp = &sR[col * CM + kb + hf * 8];
    const bf16x16 Bm = cat16(*(const bf16x8*)bp, *(const bf16x8*)(bp + 16));
    const bf16x16 Aq = cat16(*(const bf16x8*)(qRow + kb + hf * 8),
                             *(const bf16x8*)(qRow + kb + 16 + hf * 8));
    const bf16x16 At = cat16(*(const bf16x8*)(tRow + kb + hf * 8),
                             *(const bf16x8*)(tRow + kb + 16 + hf * 8));
    aa = wmma_bf16(Aq, Bm, aa);
    ab = wmma_bf16(At, Bm, ab);
  }

  const int n = n0 + col;
  const size_t ob = (size_t)b * (4 * CD) * CN + n;
#pragma unroll
  for (int r = 0; r < 8; ++r) {
    const int d = d0 + r + hf * 8;
    const float cv = C[((size_t)b * CD + d) * CN + n];
    const float av = aa[r], bv = ab[r];
    OUT[ob + (size_t)d * CN]              = cv;
    OUT[ob + (size_t)(CD + d) * CN]       = av;
    OUT[ob + (size_t)(2 * CD + d) * CN]   = cv * av;
    OUT[ob + (size_t)(3 * CD + d) * CN]   = cv * bv;
  }
}

// ---------------------------------------------------------------------------
extern "C" void kernel_launch(void* const* d_in, const int* in_sizes, int n_in,
                              void* d_out, int out_size, void* d_ws, size_t ws_size,
                              hipStream_t stream) {
  const float* q  = (const float*)d_in[0];
  const float* c  = (const float*)d_in[1];
  const float* w1 = (const float*)d_in[2];
  const float* w2 = (const float*)d_in[3];
  const float* w3 = (const float*)d_in[4];
  float* out = (float*)d_out;

  char* ws = (char*)d_ws;
  size_t o = 0;
  float*  S    = (float*)(ws + o);  o += (size_t)CB * CN * CM * 4;  // 128 MB
  __bf16* QH   = (__bf16*)(ws + o); o += (size_t)CB * CD * CM * 2;
  __bf16* QT   = (__bf16*)(ws + o); o += (size_t)CB * CM * CD * 2;
  __bf16* CH   = (__bf16*)(ws + o); o += (size_t)CB * CD * CN * 2;
  __bf16* CT   = (__bf16*)(ws + o); o += (size_t)CB * CN * CD * 2;
  __bf16* TMPH = (__bf16*)(ws + o); o += (size_t)CB * CD * CM * 2;
  float*  V1   = (float*)(ws + o);  o += (size_t)CB * CM * 4;
  float*  V2   = (float*)(ws + o);  o += (size_t)CB * CN * 4;
  float*  RMAX = (float*)(ws + o);  o += (size_t)CB * CN * 4;
  float*  RRS  = (float*)(ws + o);  o += (size_t)CB * CN * 4;
  float*  CMAX = (float*)(ws + o);  o += (size_t)CB * CM * 4;
  float*  CRS  = (float*)(ws + o);  o += (size_t)CB * CM * 4;
  float*  PM   = (float*)(ws + o);  o += (size_t)CB * 8 * CM * 4;
  float*  PS   = (float*)(ws + o);  o += (size_t)CB * 8 * CM * 4;

  // Pass 1: conversions / transposes / w-dot vectors
  k_prep<<<dim3(CM / 32, CD / 32, CB), dim3(32, 8), 0, stream>>>(q, nullptr, QH, QT, CM);
  k_prep<<<dim3(CN / 32, CD / 32, CB), dim3(32, 8), 0, stream>>>(c, w3, CH, CT, CN);
  k_rowvec<<<(CB * CM) / 256, 256, 0, stream>>>(q, w1, V1, CM);
  k_rowvec<<<(CB * CN) / 256, 256, 0, stream>>>(c, w2, V2, CN);

  // Pass 2: similarity matrix s (WMMA)
  k_sgemm<<<dim3(CN / 16, CB), 256, 0, stream>>>(CT, QT, V1, V2, S);

  // Pass 3: softmax stats
  k_rowstats<<<(CB * CN) / 8, 256, 0, stream>>>(S, RMAX, RRS);
  k_colpart<<<dim3(CM / 256, 8, CB), 256, 0, stream>>>(S, PM, PS);
  k_colmerge<<<(CB * CM) / 256, 256, 0, stream>>>(PM, PS, CMAX, CRS);

  // Pass 4: tmp = c @ sCol (WMMA)
  k_tmpgemm<<<dim3(CM / 64, CB), 256, 0, stream>>>(CH, S, CMAX, CRS, TMPH);

  // Pass 5: a, bb + fused output (WMMA)
  k_final<<<dim3(CN / 16, CB), 256, 0, stream>>>(QH, TMPH, S, RMAX, RRS, c, out);
}
